// LocalGLMnet_16501264351338
// MI455X (gfx1250) — matrix operand order repaired and checked by reading, not compiled, not run on hardware
//
#include <hip/hip_runtime.h>

// ---------------------------------------------------------------------------
// LocalGLMnet forward for MI455X (gfx1250, wave32, WMMA).
//
// Unshared 5x5 locally-connected layer cast as banded 16x16x32 f16 WMMAs:
//   out[b, i, j0+n] = sum_k A[b,k] * Bw[k,n],   K = (di in 0..4) x (cc in 0..31)
//   A[b, di*32+cc]  = xpad_f16[b, i+di, j0+cc-2]          (staged in LDS)
//   Bw[di*32+cc, n] = w[i, j0+n, di, cc-n-2] (band), else 0
// Then sigmoid + forecast/penalty reductions over i in the D-fragment layout.
// ---------------------------------------------------------------------------

typedef __attribute__((ext_vector_type(16))) _Float16 v16h;
typedef __attribute__((ext_vector_type(8)))  _Float16 v8h;
typedef __attribute__((ext_vector_type(4)))  _Float16 v4h;
typedef __attribute__((ext_vector_type(8)))  float    v8f;

#define LOOK_BACK 10
#define N_AGES    100
#define KH        5
#define KW        5
#define ETA       0.01f

// LDS geometry
#define BW_STRIDE_H   192                       // halves per Bw row (K: 5*32=160, padded)
#define BW_BYTES      (16 * BW_STRIDE_H * 2)    // 6144 B
#define ROW_STRIDE_H  128                       // halves per padded-x row (cols 0..127)
#define B_STRIDE_B    3600                      // 14*128*2 = 3584 (+16 pad => ds_load_b128 bank-free)
#define TILE_BYTES    (16 * B_STRIDE_B)         // 57600 B per wave (16 batch rows)
#define WAVES_PER_WG  4
#define SMEM_BYTES    (BW_BYTES + WAVES_PER_WG * TILE_BYTES)   // 236544 B < 320KB WGP LDS
#define N_JTILES      7                          // ceil(100/16)

__global__ void __launch_bounds__(32 * WAVES_PER_WG)
localglmnet_wmma_kernel(const float* __restrict__ x,
                        const float* __restrict__ w,
                        float* __restrict__ out)
{
    extern __shared__ char smem[];
    _Float16* bw   = (_Float16*)smem;               // banded weight matrix [16 n][192 K]
    char*     tiles = smem + BW_BYTES;

    const int tid  = threadIdx.x;
    const int lane = tid & 31;
    const int wave = tid >> 5;
    const int hi16 = (lane >> 4) & 1;               // lane half (0: lanes 0-15, 1: 16-31)
    const int l15  = lane & 15;

    char* tile = tiles + wave * TILE_BYTES;         // this wave's padded-x f16 tile
    const int b0 = blockIdx.x * (16 * WAVES_PER_WG) + wave * 16;

    // ---- one-time init: zero Bw (band complement stays 0 forever) ----
    {
        uint4 z4 = {0u, 0u, 0u, 0u};
        uint4* p = (uint4*)bw;
        for (int q = tid; q < BW_BYTES / 16; q += 32 * WAVES_PER_WG) p[q] = z4;
    }
    // ---- one-time init: zero this wave's x tile (pad rows/cols) ----
    {
        uint4 z4 = {0u, 0u, 0u, 0u};
        uint4* p = (uint4*)tile;
        for (int q = lane; q < TILE_BYTES / 16; q += 32) p[q] = z4;
    }
    // ---- stage x -> f16 padded tile: LDS[b][row+2][col+4] = x[b,row,col] ----
    {
        const float* xg = x + (long)b0 * (LOOK_BACK * N_AGES);
        // 16 rows * 1000 f32 = 4000 float4, each maps inside one (b,row)
        for (int t = lane; t < 4000; t += 32) {
            float4 v = ((const float4*)xg)[t];
            int e   = t * 4;
            int bb  = e / 1000;
            int rem = e - bb * 1000;
            int row = rem / 100;
            int col = rem - row * 100;
            v4h hv;
            hv[0] = (_Float16)v.x; hv[1] = (_Float16)v.y;
            hv[2] = (_Float16)v.z; hv[3] = (_Float16)v.w;
            *(v4h*)(tile + bb * B_STRIDE_B + (row + 2) * (ROW_STRIDE_H * 2)
                         + (col + 4) * 2) = hv;
        }
    }

    for (int jt = 0; jt < N_JTILES; ++jt) {
        const int j0 = jt * 16;
        float facc[8], pacc[8];
        #pragma unroll
        for (int k = 0; k < 8; ++k) { facc[k] = 0.0f; pacc[k] = 0.0f; }

        for (int i = 0; i < LOOK_BACK; ++i) {
            // guard Bw (WAW vs previous readers) + first-iter init visibility
            __syncthreads();
            // cooperative banded-Bw build: 80 threads, 5 band halves each
            if (tid < 80) {
                int n  = tid & 15;
                int di = tid >> 4;
                int j  = j0 + n;
                _Float16* row = bw + n * BW_STRIDE_H + di * 32 + (n + 2);
                if (j < N_AGES) {
                    const float* wp = w + (((i * N_AGES + j) * KH + di) * KW);
                    #pragma unroll
                    for (int dj = 0; dj < KW; ++dj) row[dj] = (_Float16)wp[dj];
                } else {
                    #pragma unroll
                    for (int dj = 0; dj < KW; ++dj) row[dj] = (_Float16)0.0f;
                }
            }
            __syncthreads();

            // 5 K-chunks of v_wmma_f32_16x16x32_f16, 2-way accumulator ILP
            v8f c0 = {}; v8f c1 = {};
            #pragma unroll
            for (int di = 0; di < KH; ++di) {
                // A fragment (16-bit A 16x32 layout): M = l15,
                // lanes<16: K 0..7 / 16..23 ; lanes>=16: K 8..15 / 24..31
                const char* abase = tile + l15 * B_STRIDE_B
                                  + (i + di) * (ROW_STRIDE_H * 2)
                                  + (j0 + hi16 * 8) * 2;
                v8h alo = *(const v8h*)abase;
                v8h ahi = *(const v8h*)(abase + 32);
                v16h a;
                #pragma unroll
                for (int q = 0; q < 8; ++q) { a[q] = alo[q]; a[q + 8] = ahi[q]; }

                // B fragment (16-bit B 32x16): N = l15,
                // lanes<16: K 0..7 / 16..23 ; lanes>=16: K 8..15 / 24..31
                const _Float16* bbase = bw + l15 * BW_STRIDE_H + di * 32 + hi16 * 8;
                v8h blo = *(const v8h*)bbase;
                v8h bhi = *(const v8h*)(bbase + 16);
                v16h bm;
                #pragma unroll
                for (int q = 0; q < 8; ++q) { bm[q] = blo[q]; bm[q + 8] = bhi[q]; }

                if (di & 1)
                    c1 = __builtin_amdgcn_wmma_f32_16x16x32_f16(
                             false, a, false, bm, (short)0, c1, false, false);
                else
                    c0 = __builtin_amdgcn_wmma_f32_16x16x32_f16(
                             false, a, false, bm, (short)0, c0, false, false);
            }

            // epilogue on D fragments: M = k + 8*hi16 -> b row; N = l15 -> j col
            #pragma unroll
            for (int k = 0; k < 8; ++k) {
                float z = c0[k] + c1[k];
                float s = 1.0f / (1.0f + __expf(-z));        // sigmoid
                int blocal = k + hi16 * 8;
                const _Float16* xp = (const _Float16*)(tile + blocal * B_STRIDE_B
                                       + (i + 2) * (ROW_STRIDE_H * 2))
                                       + (j0 + l15 + 4);
                float xv = (float)(*xp);                      // x[b,i,j] (f16 staged)
                facc[k] += xv * s;                            // forecast
                pacc[k] += ETA * s * s;                       // ridge penalty (alpha=0)
            }
        }

        // store: out[b, 0, j] = forecast, out[b, 1, j] = penalty
        int j = j0 + l15;
        if (j < N_AGES) {
            #pragma unroll
            for (int k = 0; k < 8; ++k) {
                int b = b0 + k + hi16 * 8;
                out[(b * 2 + 0) * N_AGES + j] = facc[k];
                out[(b * 2 + 1) * N_AGES + j] = pacc[k];
            }
        }
    }
}

extern "C" void kernel_launch(void* const* d_in, const int* in_sizes, int n_in,
                              void* d_out, int out_size, void* d_ws, size_t ws_size,
                              hipStream_t stream) {
    const float* x = (const float*)d_in[0];     // (B, 10, 100) f32
    const float* w = (const float*)d_in[1];     // (10, 100, 5, 5) f32
    float* out = (float*)d_out;                 // (B, 2, 100) f32

    const int batch = in_sizes[0] / (LOOK_BACK * N_AGES);   // 16384
    const int bt    = 16 * WAVES_PER_WG;                    // 64 batch rows / WG

    (void)hipFuncSetAttribute((const void*)localglmnet_wmma_kernel,
                              hipFuncAttributeMaxDynamicSharedMemorySize,
                              SMEM_BYTES);

    dim3 grid(batch / bt);
    dim3 block(32 * WAVES_PER_WG);
    localglmnet_wmma_kernel<<<grid, block, SMEM_BYTES, stream>>>(x, w, out);
}
